// SharedBottom_5205500363292
// MI455X (gfx1250) — matrix operand order repaired and checked by reading, not compiled, not run on hardware
//
#include <hip/hip_runtime.h>
#include <hip/hip_bf16.h>

// ---------------------------------------------------------------------------
// SharedBottom recommender on MI455X (gfx1250, wave32, WMMA).
// Pipeline: gather(emb->bf16 X) -> GEMM1 relu -> GEMM2 relu ->
//           4x towers (GEMM relu x2) -> dot64 + sigmoid + domain select.
// All matmuls use v_wmma_f32_16x16x32_bf16 with f32 accumulation.
// ---------------------------------------------------------------------------

typedef __attribute__((ext_vector_type(16))) __bf16 v16bf;
typedef __attribute__((ext_vector_type(8)))  float  v8f;

#define B_ROWS 16384
#define F_FEAT 20
#define V_SIZE 100000
#define E_DIM  32
#define DBOT   640     // F*E
#define D1     512
#define D2     256
#define T1D    128
#define T2D    64
#define NDOM   4

__device__ __forceinline__ unsigned short f32_to_bf16(float f) {
  unsigned int u = __float_as_uint(f);
  unsigned int r = u + 0x7FFFu + ((u >> 16) & 1u);   // round-to-nearest-even
  return (unsigned short)(r >> 16);
}
__device__ __forceinline__ float bf16_to_f32(unsigned short s) {
  return __uint_as_float(((unsigned int)s) << 16);
}

// ---------------------------------------------------------------------------
// Gather: X[row, f*32 + e] = emb[f, ids[row,f], e], stored bf16 row-major.
// One thread per (row, feature): 8x float4 loads, 4x 16B stores.
// ---------------------------------------------------------------------------
__global__ void gather_kernel(const int* __restrict__ ids,
                              const float* __restrict__ emb,
                              unsigned short* __restrict__ X) {
  int t = blockIdx.x * blockDim.x + threadIdx.x;
  if (t >= B_ROWS * F_FEAT) return;
  int f   = t % F_FEAT;
  int row = t / F_FEAT;
  long long id = (long long)ids[(long long)row * F_FEAT + f];
  const float* src = emb + ((long long)f * V_SIZE + id) * E_DIM;
  unsigned int pk[16];
#pragma unroll
  for (int i = 0; i < 8; ++i) {
    float4 v = ((const float4*)src)[i];
    pk[2 * i]     = (unsigned int)f32_to_bf16(v.x) | ((unsigned int)f32_to_bf16(v.y) << 16);
    pk[2 * i + 1] = (unsigned int)f32_to_bf16(v.z) | ((unsigned int)f32_to_bf16(v.w) << 16);
  }
  uint4* dst = (uint4*)(X + (long long)row * DBOT + f * E_DIM);
#pragma unroll
  for (int i = 0; i < 4; ++i)
    dst[i] = make_uint4(pk[4 * i], pk[4 * i + 1], pk[4 * i + 2], pk[4 * i + 3]);
}

// ---------------------------------------------------------------------------
// Weight repack: fp32 W[z][K][N] (row-major) -> bf16 packed B-fragments.
// B layout used here (contiguous k, same order as the A loads): VGPR j, lanes 0-15: K=2j,2j+1
// (N=lane); lanes 16-31: K=16+2j,17+2j (N=lane-16). One thread per
// (z, ktile, ntile, lane) writes 16 contiguous bf16 (32B).
// ---------------------------------------------------------------------------
__global__ void packw_kernel(const float* __restrict__ W,
                             unsigned short* __restrict__ P,
                             int K, int N, int nz) {
  int Kt = K >> 5, Nt = N >> 4;
  int total = nz * Kt * Nt * 32;
  int t = blockIdx.x * blockDim.x + threadIdx.x;
  if (t >= total) return;
  int lane = t & 31;
  int tile = t >> 5;
  int nt = tile % Nt;
  int kt = (tile / Nt) % Kt;
  int z  = tile / (Nt * Kt);
  int h = lane >> 4, r = lane & 15;
  const float* Wz = W + (long long)z * K * N;
  int kbase = kt * 32 + 16 * h;
  int nbase = nt * 16 + r;
  unsigned int pk[8];
#pragma unroll
  for (int j = 0; j < 8; ++j) {
    float w0 = Wz[(long long)(kbase + 2 * j)     * N + nbase];
    float w1 = Wz[(long long)(kbase + 2 * j + 1) * N + nbase];
    pk[j] = (unsigned int)f32_to_bf16(w0) | ((unsigned int)f32_to_bf16(w1) << 16);
  }
  uint4* out = (uint4*)(P + (long long)t * 16);
  out[0] = make_uint4(pk[0], pk[1], pk[2], pk[3]);
  out[1] = make_uint4(pk[4], pk[5], pk[6], pk[7]);
}

// ---------------------------------------------------------------------------
// BF16 GEMM + bias + ReLU.  C[z] = relu(A[z] (MxK) @ W[z] (KxN) + bias[z]).
// 256 threads = 8 waves; block tile = 128x64; wave tile = 32x32 (2x2 frags).
// A: row-major bf16 (pairs (k,k+1) contiguous -> 8x b32 loads per fragment,
//    in contiguous k order, same as packed B: VGPR i holds K = 16h+2i).
// B: pre-packed fragments -> 2x b128 loads per fragment.
// Output: row-major bf16 (feeds next layer's A operand).
// ---------------------------------------------------------------------------
__global__ void __launch_bounds__(256)
gemm_bf16_relu(const unsigned short* __restrict__ A, long long aZstride,
               const unsigned short* __restrict__ Bp, long long wZstride,
               const float* __restrict__ bias, int bZstride,
               unsigned short* __restrict__ Out, long long oZstride,
               int M, int N, int K) {
  int z = blockIdx.z;
  A    += (long long)z * aZstride;
  Bp   += (long long)z * wZstride;
  bias += (long long)z * bZstride;
  Out  += (long long)z * oZstride;

  int lane = threadIdx.x & 31;
  int w    = threadIdx.x >> 5;   // 0..7
  int wm = w & 3, wn = w >> 2;
  int h = lane >> 4, r = lane & 15;

  int rowbase = blockIdx.x * 128 + wm * 32;   // two 16-row m-tiles
  int colbase = blockIdx.y * 64  + wn * 32;   // two 16-col n-tiles
  int Nt = N >> 4;
  int Kt = K >> 5;
  int nt0 = colbase >> 4;

  union { v16bf v; unsigned int u[8]; } a0, a1, b0, b1;
  union { v8f v; float f[8]; } c00, c01, c10, c11;
  v8f zero = {0.f, 0.f, 0.f, 0.f, 0.f, 0.f, 0.f, 0.f};
  c00.v = zero; c01.v = zero; c10.v = zero; c11.v = zero;

  const unsigned short* arow0 = A + (long long)(rowbase + r)      * K;
  const unsigned short* arow1 = A + (long long)(rowbase + 16 + r) * K;

  for (int kt = 0; kt < Kt; ++kt) {
    int kb = kt * 32;
#pragma unroll
    for (int i = 0; i < 8; ++i) {
      int k = kb + h * 16 + 2 * i;
      a0.u[i] = *(const unsigned int*)(arow0 + k);
      a1.u[i] = *(const unsigned int*)(arow1 + k);
    }
    const uint4* pb0 = (const uint4*)(Bp + ((long long)(kt * Nt + nt0)     * 32 + lane) * 16);
    const uint4* pb1 = (const uint4*)(Bp + ((long long)(kt * Nt + nt0 + 1) * 32 + lane) * 16);
    ((uint4*)b0.u)[0] = pb0[0];
    ((uint4*)b0.u)[1] = pb0[1];
    ((uint4*)b1.u)[0] = pb1[0];
    ((uint4*)b1.u)[1] = pb1[1];

    c00.v = __builtin_amdgcn_wmma_f32_16x16x32_bf16(false, a0.v, false, b0.v, (short)0, c00.v, false, false);
    c01.v = __builtin_amdgcn_wmma_f32_16x16x32_bf16(false, a0.v, false, b1.v, (short)0, c01.v, false, false);
    c10.v = __builtin_amdgcn_wmma_f32_16x16x32_bf16(false, a1.v, false, b0.v, (short)0, c10.v, false, false);
    c11.v = __builtin_amdgcn_wmma_f32_16x16x32_bf16(false, a1.v, false, b1.v, (short)0, c11.v, false, false);
  }

  // Epilogue: C/D layout — VGPR i holds (M = i + 8h, N = r) per 16x16 tile.
  float bv0 = bias[colbase + r];
  float bv1 = bias[colbase + 16 + r];
  int c0 = colbase + r, c1 = colbase + 16 + r;
#pragma unroll
  for (int i = 0; i < 8; ++i) {
    int m0 = rowbase + i + 8 * h;
    int m1 = m0 + 16;
    float v00 = c00.f[i] + bv0; v00 = v00 > 0.f ? v00 : 0.f;
    float v01 = c01.f[i] + bv1; v01 = v01 > 0.f ? v01 : 0.f;
    float v10 = c10.f[i] + bv0; v10 = v10 > 0.f ? v10 : 0.f;
    float v11 = c11.f[i] + bv1; v11 = v11 > 0.f ? v11 : 0.f;
    Out[(long long)m0 * N + c0] = f32_to_bf16(v00);
    Out[(long long)m0 * N + c1] = f32_to_bf16(v01);
    Out[(long long)m1 * N + c0] = f32_to_bf16(v10);
    Out[(long long)m1 * N + c1] = f32_to_bf16(v11);
  }
}

// ---------------------------------------------------------------------------
// Final: logit = T2[dom[b], b, :] . TW3[dom[b]] + Tb3[dom[b]]; out = sigmoid.
// ---------------------------------------------------------------------------
__global__ void tower_out_kernel(const unsigned short* __restrict__ T2,
                                 const float* __restrict__ TW3,
                                 const float* __restrict__ Tb3,
                                 const int* __restrict__ dom,
                                 float* __restrict__ out) {
  int b = blockIdx.x * blockDim.x + threadIdx.x;
  if (b >= B_ROWS) return;
  int d = dom[b];
  const unsigned short* t = T2 + ((long long)d * B_ROWS + b) * T2D;
  const float* wv = TW3 + d * T2D;
  float acc = Tb3[d];
#pragma unroll
  for (int i = 0; i < T2D; ++i) acc += bf16_to_f32(t[i]) * wv[i];
  out[b] = 1.0f / (1.0f + __expf(-acc));
}

// ---------------------------------------------------------------------------
extern "C" void kernel_launch(void* const* d_in, const int* in_sizes, int n_in,
                              void* d_out, int out_size, void* d_ws, size_t ws_size,
                              hipStream_t stream) {
  const int*   feature_ids = (const int*)d_in[0];
  const int*   domain_id   = (const int*)d_in[1];
  const float* emb = (const float*)d_in[2];
  const float* W1  = (const float*)d_in[3];
  const float* b1  = (const float*)d_in[4];
  const float* W2  = (const float*)d_in[5];
  const float* b2  = (const float*)d_in[6];
  const float* TW1 = (const float*)d_in[7];
  const float* Tb1 = (const float*)d_in[8];
  const float* TW2 = (const float*)d_in[9];
  const float* Tb2 = (const float*)d_in[10];
  const float* TW3 = (const float*)d_in[11];
  const float* Tb3 = (const float*)d_in[12];
  float* out = (float*)d_out;

  char* ws = (char*)d_ws;
  size_t off = 0;
  auto wsalloc = [&](size_t bytes) -> void* {
    void* p = ws + off;
    off += (bytes + 255) & ~(size_t)255;
    return p;
  };
  unsigned short* X   = (unsigned short*)wsalloc((size_t)B_ROWS * DBOT * 2);
  unsigned short* W1p = (unsigned short*)wsalloc((size_t)DBOT * D1 * 2);
  unsigned short* W2p = (unsigned short*)wsalloc((size_t)D1 * D2 * 2);
  unsigned short* TW1p= (unsigned short*)wsalloc((size_t)NDOM * D2 * T1D * 2);
  unsigned short* TW2p= (unsigned short*)wsalloc((size_t)NDOM * T1D * T2D * 2);
  unsigned short* H1  = (unsigned short*)wsalloc((size_t)B_ROWS * D1 * 2);
  unsigned short* H2  = (unsigned short*)wsalloc((size_t)B_ROWS * D2 * 2);
  unsigned short* T1  = (unsigned short*)wsalloc((size_t)NDOM * B_ROWS * T1D * 2);
  unsigned short* T2  = (unsigned short*)wsalloc((size_t)NDOM * B_ROWS * T2D * 2);
  (void)ws_size; (void)in_sizes; (void)n_in; (void)out_size;

  // 1) Repack weights to WMMA B-fragment layout (bf16).
  {
    int tot = (DBOT / 32) * (D1 / 16) * 32;                 // W1
    packw_kernel<<<(tot + 255) / 256, 256, 0, stream>>>(W1, W1p, DBOT, D1, 1);
    tot = (D1 / 32) * (D2 / 16) * 32;                       // W2
    packw_kernel<<<(tot + 255) / 256, 256, 0, stream>>>(W2, W2p, D1, D2, 1);
    tot = NDOM * (D2 / 32) * (T1D / 16) * 32;               // TW1
    packw_kernel<<<(tot + 255) / 256, 256, 0, stream>>>(TW1, TW1p, D2, T1D, NDOM);
    tot = NDOM * (T1D / 32) * (T2D / 16) * 32;              // TW2
    packw_kernel<<<(tot + 255) / 256, 256, 0, stream>>>(TW2, TW2p, T1D, T2D, NDOM);
  }

  // 2) Embedding gather -> bf16 X (B x 640).
  {
    int tot = B_ROWS * F_FEAT;
    gather_kernel<<<(tot + 255) / 256, 256, 0, stream>>>(feature_ids, emb, X);
  }

  // 3) H1 = relu(X @ W1 + b1)   [16384 x 640 x 512]
  gemm_bf16_relu<<<dim3(B_ROWS / 128, D1 / 64, 1), 256, 0, stream>>>(
      X, 0, W1p, 0, b1, 0, H1, 0, B_ROWS, D1, DBOT);

  // 4) H2 = relu(H1 @ W2 + b2)  [16384 x 512 x 256]
  gemm_bf16_relu<<<dim3(B_ROWS / 128, D2 / 64, 1), 256, 0, stream>>>(
      H1, 0, W2p, 0, b2, 0, H2, 0, B_ROWS, D2, D1);

  // 5) T1[d] = relu(H2 @ TW1[d] + Tb1[d])  [z=4: 16384 x 256 x 128]
  gemm_bf16_relu<<<dim3(B_ROWS / 128, T1D / 64, NDOM), 256, 0, stream>>>(
      H2, 0, TW1p, (long long)D2 * T1D, Tb1, T1D,
      T1, (long long)B_ROWS * T1D, B_ROWS, T1D, D2);

  // 6) T2[d] = relu(T1[d] @ TW2[d] + Tb2[d])  [z=4: 16384 x 128 x 64]
  gemm_bf16_relu<<<dim3(B_ROWS / 128, 1, NDOM), 256, 0, stream>>>(
      T1, (long long)B_ROWS * T1D, TW2p, (long long)T1D * T2D, Tb2, T2D,
      T2, (long long)B_ROWS * T2D, B_ROWS, T2D, T1D);

  // 7) Per-row dot(64) + sigmoid + domain select.
  tower_out_kernel<<<(B_ROWS + 255) / 256, 256, 0, stream>>>(
      T2, TW3, Tb3, domain_id, out);
}
